// QKNormalizedMultiheadAttention_85899345920506
// MI455X (gfx1250) — matrix operand order repaired and checked by reading, not compiled, not run on hardware
//
#include <hip/hip_runtime.h>

// ---------------------------------------------------------------------------
// QK-normalized multi-head attention for MI455X (gfx1250), wave32 + WMMA bf16.
// E=768, H=12, D=64, B=2, S=T=2048.
//
// Pipeline:
//   0) cvt_bf16_kernel : f32 -> bf16 for weights + activations (memory bound)
//   1) qkv_proj_kernel : fused QKV GEMM + bias + L2 norm (q,k) + layout shuffle
//   2) attn_kernel     : flash attention; K/V tiles staged in LDS via
//                        GLOBAL_LOAD_ASYNC_TO_LDS_B128 (double buffered)
//   3) out_proj_kernel : output GEMM + bias, f32 store
// ---------------------------------------------------------------------------

typedef __attribute__((ext_vector_type(16))) __bf16        v16bf;
typedef __attribute__((ext_vector_type(8)))  float         v8f;
typedef __attribute__((ext_vector_type(8)))  unsigned int  v8u;
typedef __attribute__((ext_vector_type(4)))  unsigned int  v4u;
typedef __attribute__((ext_vector_type(4)))  int           v4i;

typedef __attribute__((address_space(1))) v4i* gv4i_ptr;   // global v4i*
typedef __attribute__((address_space(3))) v4i* lv4i_ptr;   // LDS v4i*

#define NE        768
#define NH        12
#define ND        64
#define NB        2
#define NS        2048
#define LOG2E     1.44269504088896f
#define QK_SCALE  0.125f                 /* D^-0.5 = 1/8 */

__device__ __forceinline__ unsigned short f2bf_bits(float f) {
    unsigned u = __float_as_uint(f);
    unsigned r = (u + 0x7FFFu + ((u >> 16) & 1u)) >> 16;   // round-to-nearest-even
    return (unsigned short)r;
}
__device__ __forceinline__ unsigned pack2(float a, float b) {
    return (unsigned)f2bf_bits(a) | ((unsigned)f2bf_bits(b) << 16);
}

// A fragment (16x32 bf16): lane m holds row m; per-lane data = K in
// [hf*16, hf*16+16)  (hf = lane/16).
__device__ __forceinline__ v16bf load_a_bf16(const unsigned short* __restrict__ row,
                                             int k0, int hf) {
    v4u lo = *(const v4u*)(row + k0 + hf * 16);
    v4u hi = *(const v4u*)(row + k0 + hf * 16 + 8);
    v8u u;
    u[0] = lo[0]; u[1] = lo[1]; u[2] = lo[2]; u[3] = lo[3];
    u[4] = hi[0]; u[5] = hi[1]; u[6] = hi[2]; u[7] = hi[3];
    return __builtin_bit_cast(v16bf, u);
}

// B fragment (32x16 bf16): lane (n = lane&15) holds 16 contiguous K values
// starting at K = hf*16; caller passes the per-lane base pointer.
__device__ __forceinline__ v16bf load_b_bf16(const unsigned short* p) {
    v4u lo = *(const v4u*)(p);
    v4u hi = *(const v4u*)(p + 8);
    v8u u;
    u[0] = lo[0]; u[1] = lo[1]; u[2] = lo[2]; u[3] = lo[3];
    u[4] = hi[0]; u[5] = hi[1]; u[6] = hi[2]; u[7] = hi[3];
    return __builtin_bit_cast(v16bf, u);
}

__device__ __forceinline__ v8f wmma_bf16(v16bf a, v16bf b, v8f c) {
    return __builtin_amdgcn_wmma_f32_16x16x32_bf16(false, a, false, b, (short)0, c,
                                                   false, false);
}

// ---- CDNA5 async global->LDS copy (ASYNCcnt-tracked), 16B per lane ---------
__device__ __forceinline__ void async_b128(const unsigned short* gsrc,
                                           unsigned short* lds) {
#if __has_builtin(__builtin_amdgcn_global_load_async_to_lds_b128)
    __builtin_amdgcn_global_load_async_to_lds_b128(
        (gv4i_ptr)(unsigned short*)gsrc, (lv4i_ptr)lds, 0, 0);
#else
    unsigned lofs = (unsigned)(unsigned long long)(lv4i_ptr)lds;
    asm volatile("global_load_async_to_lds_b128 %0, %1, off"
                 :: "v"(lofs), "v"(gsrc) : "memory");
#endif
}
__device__ __forceinline__ void wait_async_le4() {
#if __has_builtin(__builtin_amdgcn_s_wait_asynccnt)
    __builtin_amdgcn_s_wait_asynccnt(4);
#else
    asm volatile("s_wait_asynccnt 0x4" ::: "memory");
#endif
}
__device__ __forceinline__ void wait_async_0() {
#if __has_builtin(__builtin_amdgcn_s_wait_asynccnt)
    __builtin_amdgcn_s_wait_asynccnt(0);
#else
    asm volatile("s_wait_asynccnt 0x0" ::: "memory");
#endif
}

// ---------------------------------------------------------------------------
// Kernel 0: bulk f32 -> bf16 conversion (grid-stride over float4 chunks).
// ---------------------------------------------------------------------------
__global__ __launch_bounds__(256) void cvt_bf16_kernel(
    const float* __restrict__ src, unsigned short* __restrict__ dst, int n4)
{
    int i = blockIdx.x * blockDim.x + threadIdx.x;
    int stride = gridDim.x * blockDim.x;
    for (; i < n4; i += stride) {
        const float4* p = (const float4*)src + 2 * i;
        float4 a = p[0], b = p[1];
        v4u u;
        u[0] = pack2(a.x, a.y); u[1] = pack2(a.z, a.w);
        u[2] = pack2(b.x, b.y); u[3] = pack2(b.z, b.w);
        *((v4u*)dst + i) = u;                  // 8 bf16 per iteration, 16B store
    }
}

// ---------------------------------------------------------------------------
// Kernel 1: fused QKV projection (+bias, +L2 norm for q/k, layout shuffles).
// grid = (E/64 = heads, (B*S)/64, 3), block = 128 (4 waves, 16 rows/wave).
// ---------------------------------------------------------------------------
__global__ __launch_bounds__(128) void qkv_proj_kernel(
    const unsigned short* __restrict__ Xq, const unsigned short* __restrict__ Xk,
    const unsigned short* __restrict__ Xv,
    const unsigned short* __restrict__ Wq, const float* __restrict__ bq,
    const unsigned short* __restrict__ Wk, const float* __restrict__ bk,
    const unsigned short* __restrict__ Wv, const float* __restrict__ bv,
    unsigned short* __restrict__ qn, unsigned short* __restrict__ kn,
    unsigned short* __restrict__ vt)
{
    const int lane = threadIdx.x & 31;
    const int wv   = threadIdx.x >> 5;
    const int hf   = lane >> 4;
    const int l16  = lane & 15;
    const int which = blockIdx.z;                 // 0=q 1=k 2=v
    const int h     = blockIdx.x;                 // head == 64-col block
    const int m0    = blockIdx.y * 64 + wv * 16;  // first M row of this wave

    const unsigned short* X = (which == 0) ? Xq : ((which == 1) ? Xk : Xv);
    const unsigned short* W = (which == 0) ? Wq : ((which == 1) ? Wk : Wv);
    const float* bias       = (which == 0) ? bq : ((which == 1) ? bk : bv);

    const unsigned short* arow = X + (size_t)(m0 + l16) * NE;
    const unsigned short* brow[4];
#pragma unroll
    for (int j = 0; j < 4; ++j)
        brow[j] = W + (size_t)(h * 64 + j * 16 + l16) * NE + hf * 16;

    v8f acc[4] = {};
    for (int k0 = 0; k0 < NE; k0 += 32) {
        __builtin_prefetch(arow + k0 + 64, 0, 0);          // global_prefetch_b8
        v16bf a = load_a_bf16(arow, k0, hf);
#pragma unroll
        for (int j = 0; j < 4; ++j)
            acc[j] = wmma_bf16(a, load_b_bf16(brow[j] + k0), acc[j]);
    }

    float bb[4];
#pragma unroll
    for (int j = 0; j < 4; ++j) bb[j] = bias[h * 64 + j * 16 + l16];
#pragma unroll
    for (int j = 0; j < 4; ++j)
#pragma unroll
        for (int r = 0; r < 8; ++r) acc[j][r] += bb[j];

    if (which < 2) {
        // per-row L2 norm over the head's 64 columns; fold SCALE*log2e into q
        const float scl = (which == 0) ? (QK_SCALE * LOG2E) : 1.0f;
        unsigned short* dst = (which == 0) ? qn : kn;
#pragma unroll
        for (int r = 0; r < 8; ++r) {
            float ss = 0.f;
#pragma unroll
            for (int j = 0; j < 4; ++j) ss += acc[j][r] * acc[j][r];
#pragma unroll
            for (int m = 1; m < 16; m <<= 1) ss += __shfl_xor(ss, m, 32);
            float inv = scl / fmaxf(sqrtf(ss), 1e-6f);
            int row = m0 + r + hf * 8;
            int b = row >> 11, t = row & 2047;             // S = 2048
            size_t base = ((size_t)(b * NH + h) * NS + t) * ND;
#pragma unroll
            for (int j = 0; j < 4; ++j)
                dst[base + j * 16 + l16] = f2bf_bits(acc[j][r] * inv);
        }
    } else {
        // V stored transposed: vt[(b,h), d, s] so P@V B-fragments are contiguous
#pragma unroll
        for (int r = 0; r < 8; ++r) {
            int row = m0 + r + hf * 8;
            int b = row >> 11, t = row & 2047;
            size_t base = (size_t)(b * NH + h) * ND * NS;
#pragma unroll
            for (int j = 0; j < 4; ++j)
                vt[base + (size_t)(j * 16 + l16) * NS + t] = f2bf_bits(acc[j][r]);
        }
    }
}

// ---------------------------------------------------------------------------
// Kernel 2: flash attention with async-LDS staged K/V tiles (double buffered).
// grid = (T/64, H, B), block = 128; each wave owns 16 query rows x D=64.
// ---------------------------------------------------------------------------
__global__ __launch_bounds__(128) void attn_kernel(
    const unsigned short* __restrict__ qn, const unsigned short* __restrict__ kn,
    const unsigned short* __restrict__ vt, unsigned short* __restrict__ ctx)
{
    __shared__ __align__(16) unsigned short kbuf[2][32][64];  // [s_rel][d]
    __shared__ __align__(16) unsigned short vbuf[2][64][32];  // [d][s_rel]
    __shared__ __align__(16) unsigned short pbuf[4][16][32];  // per-wave P tile

    const int tid  = threadIdx.x;
    const int lane = tid & 31;
    const int wv   = tid >> 5;
    const int hf   = lane >> 4;
    const int l16  = lane & 15;
    const int h = blockIdx.y, b = blockIdx.z;
    const int t0 = blockIdx.x * 64 + wv * 16;
    const size_t bh = (size_t)(b * NH + h);

    const unsigned short* kbase = kn + bh * NS * ND;  // [s][d]
    const unsigned short* vbase = vt + bh * ND * NS;  // [d][s]

    // issue one 32-key K tile (4KB) + V tile (4KB) as 4 async b128 ops / lane
    auto issue_tile = [&](int bufi, int s0) {
#pragma unroll
        for (int k = 0; k < 2; ++k) {
            int c = tid + k * 128;                 // 256 chunks of 16B (K tile)
            int row = c >> 3, col = (c & 7) * 8;   // 8 chunks per 64-elem row
            async_b128(kbase + (size_t)(s0 + row) * ND + col, &kbuf[bufi][row][col]);
        }
#pragma unroll
        for (int k = 0; k < 2; ++k) {
            int c = tid + k * 128;                 // 256 chunks of 16B (V tile)
            int row = c >> 2, col = (c & 3) * 8;   // 4 chunks per 32-elem row
            async_b128(vbase + (size_t)row * NS + s0 + col, &vbuf[bufi][row][col]);
        }
    };

    // Q A-fragments (held across the whole KV sweep)
    const unsigned short* qrow = qn + (bh * NS + t0 + l16) * ND;
    v16bf aq0 = load_a_bf16(qrow, 0, hf);
    v16bf aq1 = load_a_bf16(qrow, 32, hf);

    v8f   cacc[4] = {};
    float run_max[8], run_sum[8];
#pragma unroll
    for (int r = 0; r < 8; ++r) { run_max[r] = -3.0e38f; run_sum[r] = 0.f; }

    issue_tile(0, 0);

    for (int s0 = 0; s0 < NS; s0 += 32) {
        const int cur = (s0 >> 5) & 1;
        const bool has_next = (s0 + 32) < NS;
        if (has_next) {
            issue_tile(cur ^ 1, s0 + 32);   // overlap next fill with compute
            wait_async_le4();               // in-order: current tile resident
        } else {
            wait_async_0();
        }
        __syncthreads();                    // all waves' portions visible

        // ---- scores: 16 x 32 block = two 16x16 C tiles (K from LDS) ----
        v8f sc[2] = {};
#pragma unroll
        for (int j = 0; j < 2; ++j) {
            const unsigned short* krow = &kbuf[cur][j * 16 + l16][hf * 16];
            sc[j] = wmma_bf16(aq0, load_b_bf16(krow), sc[j]);
            sc[j] = wmma_bf16(aq1, load_b_bf16(krow + 32), sc[j]);
        }

        // ---- online softmax (per row r; row values live in 16 lanes) ----
        float alpha[8];
#pragma unroll
        for (int r = 0; r < 8; ++r) {
            float bm = fmaxf(sc[0][r], sc[1][r]);
#pragma unroll
            for (int m = 1; m < 16; m <<= 1) bm = fmaxf(bm, __shfl_xor(bm, m, 32));
            float nm = fmaxf(run_max[r], bm);
            alpha[r] = exp2f(run_max[r] - nm);
            run_max[r] = nm;
            float p0 = exp2f(sc[0][r] - nm);
            float p1 = exp2f(sc[1][r] - nm);
            sc[0][r] = p0; sc[1][r] = p1;
            float bs = p0 + p1;
#pragma unroll
            for (int m = 1; m < 16; m <<= 1) bs += __shfl_xor(bs, m, 32);
            run_sum[r] = run_sum[r] * alpha[r] + bs;
        }
#pragma unroll
        for (int j2 = 0; j2 < 4; ++j2)
#pragma unroll
            for (int r = 0; r < 8; ++r) cacc[j2][r] *= alpha[r];

        // ---- C-layout -> A-fragment transpose of P through LDS ----
#pragma unroll
        for (int r = 0; r < 8; ++r) {
            int row = r + hf * 8;
            pbuf[wv][row][l16]      = f2bf_bits(sc[0][r]);
            pbuf[wv][row][16 + l16] = f2bf_bits(sc[1][r]);
        }
        __syncthreads();
        v16bf ap;
        {
            const unsigned short* prow = &pbuf[wv][l16][0];
            v4u lo = *(const v4u*)(prow + hf * 16);
            v4u hi = *(const v4u*)(prow + hf * 16 + 8);
            v8u u;
            u[0] = lo[0]; u[1] = lo[1]; u[2] = lo[2]; u[3] = lo[3];
            u[4] = hi[0]; u[5] = hi[1]; u[6] = hi[2]; u[7] = hi[3];
            ap = __builtin_bit_cast(v16bf, u);
        }

        // ---- ctx += P @ V  (V from LDS, pre-transposed [d][s]) ----
#pragma unroll
        for (int j2 = 0; j2 < 4; ++j2) {
            const unsigned short* vrow = &vbuf[cur][j2 * 16 + l16][hf * 16];
            cacc[j2] = wmma_bf16(ap, load_b_bf16(vrow), cacc[j2]);
        }
        __syncthreads();   // protect kbuf/vbuf[cur] before next iter's refill
    }

    // ---- normalize and store ctx (bf16, [B, T, E] with head offset) ----
#pragma unroll
    for (int r = 0; r < 8; ++r) {
        float inv = 1.0f / run_sum[r];
        int row = t0 + r + hf * 8;
        size_t obase = ((size_t)b * NS + row) * NE + h * 64;
#pragma unroll
        for (int j2 = 0; j2 < 4; ++j2)
            ctx[obase + j2 * 16 + l16] = f2bf_bits(cacc[j2][r] * inv);
    }
}

// ---------------------------------------------------------------------------
// Kernel 3: output projection out = ctx @ Wo^T + bo (f32 output).
// grid = (E/64, (B*T)/64), block = 128.
// ---------------------------------------------------------------------------
__global__ __launch_bounds__(128) void out_proj_kernel(
    const unsigned short* __restrict__ ctx, const unsigned short* __restrict__ Wo,
    const float* __restrict__ bo, float* __restrict__ out)
{
    const int lane = threadIdx.x & 31;
    const int wv   = threadIdx.x >> 5;
    const int hf   = lane >> 4;
    const int l16  = lane & 15;
    const int n0 = blockIdx.x * 64;
    const int m0 = blockIdx.y * 64 + wv * 16;

    const unsigned short* arow = ctx + (size_t)(m0 + l16) * NE;
    const unsigned short* brow[4];
#pragma unroll
    for (int j = 0; j < 4; ++j)
        brow[j] = Wo + (size_t)(n0 + j * 16 + l16) * NE + hf * 16;

    v8f acc[4] = {};
    for (int k0 = 0; k0 < NE; k0 += 32) {
        __builtin_prefetch(arow + k0 + 64, 0, 0);          // global_prefetch_b8
        v16bf a = load_a_bf16(arow, k0, hf);
#pragma unroll
        for (int j = 0; j < 4; ++j)
            acc[j] = wmma_bf16(a, load_b_bf16(brow[j] + k0), acc[j]);
    }

    float bb[4];
#pragma unroll
    for (int j = 0; j < 4; ++j) bb[j] = bo[n0 + j * 16 + l16];
#pragma unroll
    for (int r = 0; r < 8; ++r) {
        int row = m0 + r + hf * 8;
#pragma unroll
        for (int j = 0; j < 4; ++j)
            out[(size_t)row * NE + n0 + j * 16 + l16] = acc[j][r] + bb[j];
    }
}

// ---------------------------------------------------------------------------
extern "C" void kernel_launch(void* const* d_in, const int* in_sizes, int n_in,
                              void* d_out, int out_size, void* d_ws, size_t ws_size,
                              hipStream_t stream) {
    const float* query = (const float*)d_in[0];
    const float* key_  = (const float*)d_in[1];
    const float* value = (const float*)d_in[2];
    const float* Wq = (const float*)d_in[3];
    const float* bq = (const float*)d_in[4];
    const float* Wk = (const float*)d_in[5];
    const float* bk = (const float*)d_in[6];
    const float* Wv = (const float*)d_in[7];
    const float* bv = (const float*)d_in[8];
    const float* Wo = (const float*)d_in[9];
    const float* bo = (const float*)d_in[10];
    float* out = (float*)d_out;

    // workspace layout (ushort units)
    const size_t nAct = (size_t)NB * NS * NE;   // 3,145,728
    const size_t nW   = (size_t)NE * NE;        //   589,824
    unsigned short* qn   = (unsigned short*)d_ws;
    unsigned short* kn   = qn + nAct;
    unsigned short* vt   = kn + nAct;
    unsigned short* ctx  = vt + nAct;
    unsigned short* xq_b = ctx + nAct;
    unsigned short* xk_b = xq_b + nAct;
    unsigned short* xv_b = xk_b + nAct;
    unsigned short* wq_b = xv_b + nAct;
    unsigned short* wk_b = wq_b + nW;
    unsigned short* wv_b = wk_b + nW;
    unsigned short* wo_b = wv_b + nW;
    // total = 7*nAct + 4*nW = 24,379,392 ushorts = 48.8 MB

    // ---- 0) convert weights + activations to bf16 (memory bound) ----
    {
        dim3 cb(256);
        int nAct8 = (int)(nAct / 8), nW8 = (int)(nW / 8);
        dim3 ga((nAct8 + 255) / 256), gw((nW8 + 255) / 256);
        cvt_bf16_kernel<<<ga, cb, 0, stream>>>(query, xq_b, nAct8);
        cvt_bf16_kernel<<<ga, cb, 0, stream>>>(key_,  xk_b, nAct8);
        cvt_bf16_kernel<<<ga, cb, 0, stream>>>(value, xv_b, nAct8);
        cvt_bf16_kernel<<<gw, cb, 0, stream>>>(Wq, wq_b, nW8);
        cvt_bf16_kernel<<<gw, cb, 0, stream>>>(Wk, wk_b, nW8);
        cvt_bf16_kernel<<<gw, cb, 0, stream>>>(Wv, wv_b, nW8);
        cvt_bf16_kernel<<<gw, cb, 0, stream>>>(Wo, wo_b, nW8);
    }

    dim3 blk(128);
    qkv_proj_kernel<<<dim3(NE / 64, (NB * NS) / 64, 3), blk, 0, stream>>>(
        xq_b, xk_b, xv_b, wq_b, bq, wk_b, bk, wv_b, bv, qn, kn, vt);
    attn_kernel<<<dim3(NS / 64, NH, NB), blk, 0, stream>>>(qn, kn, vt, ctx);
    out_proj_kernel<<<dim3(NE / 64, (NB * NS) / 64, 1), blk, 0, stream>>>(
        ctx, wo_b, bo, out);
}
